// VCWEModel_65231963292410
// MI455X (gfx1250) — matrix operand order repaired and checked by reading, not compiled, not run on hardware
//
#include <hip/hip_runtime.h>
#include <hip/hip_bf16.h>
#include <math.h>

// ---------------------------------------------------------------- types
typedef _Float16 v16h __attribute__((ext_vector_type(16)));
typedef _Float16 v8h  __attribute__((ext_vector_type(8)));
typedef float    v8f  __attribute__((ext_vector_type(8)));

// ---------------------------------------------------------------- sizes
#define EMBV   50000
#define CHARS  2048
#define E      128
#define BATCH  1024
#define W      6
#define CPW    5
#define NEG    5
#define NOISE  512           // B//2
#define NW     11264         // 6144 pos + 5120 neg words
#define TN     56320         // 5 * NW
#define BN_EPS 1e-5f

__device__ __forceinline__ float sigmf(float x){ return 1.f/(1.f+expf(-x)); }
__device__ __forceinline__ float clip10(float x){ return fminf(fmaxf(x,-10.f),10.f); }

// ============================================================ elementwise
__global__ void k_cast_f16(const float* __restrict__ s, _Float16* __restrict__ d, int n){
  int t = blockIdx.x*256 + threadIdx.x; if (t < n) d[t] = (_Float16)s[t];
}
__global__ void k_vecadd(const float* __restrict__ a, const float* __restrict__ b,
                         float* __restrict__ o, int n){
  int t = blockIdx.x*256 + threadIdx.x; if (t < n) o[t] = a[t] + b[t];
}
__global__ void k_zero_f32(float* p, int n){
  int t = blockIdx.x*256 + threadIdx.x; if (t < n) p[t] = 0.f;
}
__global__ void k_zero_f16(_Float16* p, int n){
  int t = blockIdx.x*256 + threadIdx.x; if (t < n) p[t] = (_Float16)0.f;
}

// ============================================================ generic WMMA GEMM
// C[M, N] = act( A[M,K](f16) * B[N,K](f16)^T + bias[n] )
// grid = (M/16, N/64), block = 32 (one wave). Each wave: 16x64 output panel,
// 4 accumulators sharing one A fragment (4x A reuse, 4 WMMA per K-step).
// K % 32 == 0, lda/ldb % 16 == 0, N % 64 == 0.
__global__ void k_gemm_wmma(const _Float16* __restrict__ A, int lda,
                            const _Float16* __restrict__ B, int ldb,
                            float* __restrict__ C, int ldc, int K,
                            const float* __restrict__ bias, int act)
{
  const int lane = threadIdx.x;
  const int m0 = blockIdx.x << 4;
  const int n0 = blockIdx.y << 6;            // 64-wide N panel
  const int r  = lane & 15;                  // A row / B row (=N col) within tile
  const int hi = lane >> 4;                  // half-wave select
  const int koffA = hi << 3;                 // 0 or 8
  const int koffB = hi << 4;                 // 0 or 16
  const _Float16* Ap = A + (size_t)(m0 + r) * lda;
  const _Float16* Bp = B + (size_t)(n0 + r) * ldb;
  const size_t bstep = (size_t)16 * ldb;
  v8f acc0 = {}, acc1 = {}, acc2 = {}, acc3 = {};
  for (int k = 0; k < K; k += 32) {
    v8h alo = *(const v8h*)(Ap + k + koffB);
    v8h ahi = *(const v8h*)(Ap + k + koffB + 8);
    v16h a;
#pragma unroll
    for (int i = 0; i < 8; ++i) { a[i] = alo[i]; a[8+i] = ahi[i]; }
    v16h b0 = *(const v16h*)(Bp             + k + koffB);
    v16h b1 = *(const v16h*)(Bp +   bstep   + k + koffB);
    v16h b2 = *(const v16h*)(Bp + 2*bstep   + k + koffB);
    v16h b3 = *(const v16h*)(Bp + 3*bstep   + k + koffB);
    acc0 = __builtin_amdgcn_wmma_f32_16x16x32_f16(false, a, false, b0, (short)0, acc0, false, false);
    acc1 = __builtin_amdgcn_wmma_f32_16x16x32_f16(false, a, false, b1, (short)0, acc1, false, false);
    acc2 = __builtin_amdgcn_wmma_f32_16x16x32_f16(false, a, false, b2, (short)0, acc2, false, false);
    acc3 = __builtin_amdgcn_wmma_f32_16x16x32_f16(false, a, false, b3, (short)0, acc3, false, false);
  }
#pragma unroll
  for (int t = 0; t < 4; ++t) {
    v8f acc = (t == 0) ? acc0 : (t == 1) ? acc1 : (t == 2) ? acc2 : acc3;
    const int nc = n0 + (t << 4) + r;
    const float bv = bias ? bias[nc] : 0.f;
#pragma unroll
    for (int v = 0; v < 8; ++v) {
      float x = acc[v] + bv;
      if (act == 1)      x = fmaxf(x, 0.f);
      else if (act == 2) x = tanhf(x);
      C[(size_t)(m0 + v + (hi << 3)) * ldc + nc] = x;
    }
  }
}

// ============================================================ conv1 (VALU, two-pass BN)
// pass 1: per-channel sum/sumsq of conv1+bias (recompute, no storage)
__global__ void k_conv1_stats(const float* __restrict__ img, const float* __restrict__ w,
                              const float* __restrict__ bias,
                              float* __restrict__ sums, float* __restrict__ sqs)
{
  const int plane = blockIdx.x;          // n*32 + c
  const int n = plane >> 5, c = plane & 31;
  __shared__ float sw[9];
  __shared__ float r1[256], r2[256];
  if (threadIdx.x < 9) sw[threadIdx.x] = w[c*9 + threadIdx.x];
  __syncthreads();
  const float bc = bias[c];
  float s = 0.f, q = 0.f;
  for (int idx = threadIdx.x; idx < 36*36; idx += 256) {
    int y = idx / 36, x = idx - y*36;
    const float* ip = img + (size_t)n*38*38 + y*38 + x;
    float acc = bc;
#pragma unroll
    for (int dy = 0; dy < 3; ++dy)
#pragma unroll
      for (int dx = 0; dx < 3; ++dx)
        acc += ip[dy*38+dx] * sw[dy*3+dx];
    s += acc; q += acc*acc;
  }
  r1[threadIdx.x] = s; r2[threadIdx.x] = q; __syncthreads();
  for (int off = 128; off > 0; off >>= 1) {
    if (threadIdx.x < off) { r1[threadIdx.x]+=r1[threadIdx.x+off]; r2[threadIdx.x]+=r2[threadIdx.x+off]; }
    __syncthreads();
  }
  if (threadIdx.x == 0) { atomicAdd(&sums[c], r1[0]); atomicAdd(&sqs[c], r2[0]); }
}

// pass 2: recompute conv, normalize, 2x2 maxpool -> pool1 [2048,32,18,18]
__global__ void k_bn1_pool(const float* __restrict__ img, const float* __restrict__ w,
                           const float* __restrict__ bias,
                           const float* __restrict__ sums, const float* __restrict__ sqs,
                           const float* __restrict__ g, const float* __restrict__ bt,
                           float* __restrict__ out)
{
  const size_t total = (size_t)2048*32*18*18;
  size_t t = (size_t)blockIdx.x*256 + threadIdx.x;
  if (t >= total) return;
  int px = (int)(t % 18); size_t t2 = t / 18;
  int py = (int)(t2 % 18); size_t t3 = t2 / 18;
  int c = (int)(t3 & 31); int n = (int)(t3 >> 5);
  const float cnt = 2048.f * 1296.f;
  float mean = sums[c] / cnt;
  float var  = sqs[c] / cnt - mean*mean;
  float scale = g[c] * rsqrtf(var + BN_EPS);
  float shift = bt[c] - mean * scale;
  const float bc = bias[c];
  float m = -3.0e38f;
#pragma unroll
  for (int dy = 0; dy < 2; ++dy)
#pragma unroll
    for (int dx = 0; dx < 2; ++dx) {
      int y = 2*py + dy, x = 2*px + dx;
      const float* ip = img + (size_t)n*38*38 + y*38 + x;
      float acc = bc;
#pragma unroll
      for (int ky = 0; ky < 3; ++ky)
#pragma unroll
        for (int kx = 0; kx < 3; ++kx)
          acc += ip[ky*38+kx] * w[c*9 + ky*3 + kx];
      m = fmaxf(m, acc*scale + shift);
    }
  out[t] = m;
}

// ============================================================ conv2: implicit-GEMM WMMA
// grid = 32768 (16 px each), block = 32. K = 288 = 9*32.
__global__ void k_conv2_wmma(const float* __restrict__ pool1, const _Float16* __restrict__ w2,
                             const float* __restrict__ bias, float* __restrict__ out)
{
  __shared__ _Float16 As[512];                 // 16 px x 32 k
  const int lane = threadIdx.x;
  const int p0 = blockIdx.x << 4;
  const int img = p0 >> 8;
  const int oy  = (p0 & 255) >> 4;             // block covers one output row
  const int r   = lane & 15;
  const int hi  = lane >> 4;
  const int koffA = hi << 3, koffB = hi << 4;
  v8f acc0 = {}, acc1 = {};
  for (int kc = 0; kc < 9; ++kc) {
    __syncthreads();
    for (int q = lane; q < 512; q += 32) {
      int m = q >> 5, j = q & 31;
      int k = kc*32 + j;
      int cin = k / 9; int rem = k - cin*9;
      int dy = rem / 3; int dx = rem - dy*3;
      float v = pool1[(((size_t)img*32 + cin)*18 + (oy+dy))*18 + (m+dx)];
      As[(m<<5) | j] = (_Float16)v;
    }
    __syncthreads();
    v16h a;
#pragma unroll
    for (int i = 0; i < 8; ++i) { a[i] = As[(r<<5) + koffB + i]; a[8+i] = As[(r<<5) + koffB + 8 + i]; }
    v16h b0 = *(const v16h*)(w2 + (size_t)r      *288 + kc*32 + koffB);
    v16h b1 = *(const v16h*)(w2 + (size_t)(r+16) *288 + kc*32 + koffB);
    acc0 = __builtin_amdgcn_wmma_f32_16x16x32_f16(false, a, false, b0, (short)0, acc0, false, false);
    acc1 = __builtin_amdgcn_wmma_f32_16x16x32_f16(false, a, false, b1, (short)0, acc1, false, false);
  }
#pragma unroll
  for (int v = 0; v < 8; ++v) {
    int ox = v + (hi << 3);
    out[(((size_t)img*32 + r)*16      + oy)*16 + ox] = acc0[v] + bias[r];
    out[(((size_t)img*32 + r + 16)*16 + oy)*16 + ox] = acc1[v] + bias[r+16];
  }
}

// per (n,c) plane of 256 elems -> channel sums
__global__ void k_plane_stats256(const float* __restrict__ x,
                                 float* __restrict__ sums, float* __restrict__ sqs)
{
  const int plane = blockIdx.x;  const int c = plane & 31;
  __shared__ float r1[256], r2[256];
  float v = x[(size_t)plane*256 + threadIdx.x];
  r1[threadIdx.x] = v; r2[threadIdx.x] = v*v; __syncthreads();
  for (int off = 128; off > 0; off >>= 1) {
    if (threadIdx.x < off) { r1[threadIdx.x]+=r1[threadIdx.x+off]; r2[threadIdx.x]+=r2[threadIdx.x+off]; }
    __syncthreads();
  }
  if (threadIdx.x == 0) { atomicAdd(&sums[c], r1[0]); atomicAdd(&sqs[c], r2[0]); }
}

// bn2 + pool -> f16 FC input [2048, 2048]
__global__ void k_bn2_pool(const float* __restrict__ conv2, const float* __restrict__ sums,
                           const float* __restrict__ sqs, const float* __restrict__ g,
                           const float* __restrict__ bt, _Float16* __restrict__ out)
{
  size_t t = (size_t)blockIdx.x*256 + threadIdx.x;
  if (t >= (size_t)2048*32*8*8) return;
  int px = (int)(t & 7), py = (int)((t >> 3) & 7);
  int c  = (int)((t >> 6) & 31), n = (int)(t >> 11);
  const float cnt = 2048.f * 256.f;
  float mean = sums[c] / cnt;
  float var  = sqs[c] / cnt - mean*mean;
  float scale = g[c] * rsqrtf(var + BN_EPS);
  float shift = bt[c] - mean * scale;
  float m = -3.0e38f;
#pragma unroll
  for (int dy = 0; dy < 2; ++dy)
#pragma unroll
    for (int dx = 0; dx < 2; ++dx)
      m = fmaxf(m, conv2[(((size_t)n*32 + c)*16 + 2*py+dy)*16 + 2*px+dx]*scale + shift);
  out[(size_t)n*2048 + c*64 + py*8 + px] = (_Float16)m;
}

// bn1d stats over 2048 rows, one block per feature column
__global__ void k_bn3_stats(const float* __restrict__ x, float* __restrict__ sums,
                            float* __restrict__ sqs)
{
  const int j = blockIdx.x;
  __shared__ float r1[256], r2[256];
  float s = 0.f, q = 0.f;
  for (int r = threadIdx.x; r < 2048; r += 256) { float v = x[(size_t)r*128 + j]; s += v; q += v*v; }
  r1[threadIdx.x] = s; r2[threadIdx.x] = q; __syncthreads();
  for (int off = 128; off > 0; off >>= 1) {
    if (threadIdx.x < off) { r1[threadIdx.x]+=r1[threadIdx.x+off]; r2[threadIdx.x]+=r2[threadIdx.x+off]; }
    __syncthreads();
  }
  if (threadIdx.x == 0) { sums[j] = r1[0]; sqs[j] = r2[0]; }
}

__global__ void k_bn3_relu_f16(const float* __restrict__ x, const float* __restrict__ sums,
                               const float* __restrict__ sqs, const float* __restrict__ g,
                               const float* __restrict__ bt, _Float16* __restrict__ out)
{
  int t = blockIdx.x*256 + threadIdx.x;
  if (t >= 2048*128) return;
  int j = t & 127;
  float mean = sums[j] / 2048.f;
  float var  = sqs[j] / 2048.f - mean*mean;
  float v = (x[t] - mean) * rsqrtf(var + BN_EPS) * g[j] + bt[j];
  out[t] = (_Float16)fmaxf(v, 0.f);
}

// ============================================================ embeddings / mining
__global__ void k_gather_u(const float* __restrict__ u_emb, const int* __restrict__ pos_u,
                           float* __restrict__ out)
{
  int t = blockIdx.x*256 + threadIdx.x; if (t >= BATCH*E) return;
  out[t] = u_emb[(size_t)pos_u[t >> 7]*E + (t & 127)];
}
__global__ void k_gather_v_mean(const float* __restrict__ v_emb, const int* __restrict__ pos_v,
                                float* __restrict__ out)
{
  int t = blockIdx.x*256 + threadIdx.x; if (t >= BATCH*E) return;
  int b = t >> 7, e = t & 127;
  float s = 0.f;
#pragma unroll
  for (int w = 0; w < W; ++w) s += v_emb[(size_t)pos_v[b*W + w]*E + e];
  out[t] = s * (1.f/6.f);
}
__global__ void k_cos(const float* __restrict__ emb_v, const float* __restrict__ v_emb,
                      const int* __restrict__ noise, float* __restrict__ cosb)
{
  int t = blockIdx.x*256 + threadIdx.x; if (t >= BATCH*NOISE) return;
  int b = t >> 9, m = t & 511;
  const float* a = emb_v + (size_t)b*E;
  const float* c = v_emb + (size_t)noise[b*NOISE + m]*E;
  float d = 0.f, na = 0.f, nb = 0.f;
  for (int e = 0; e < E; ++e) { float x = a[e], y = c[e]; d += x*y; na += x*x; nb += y*y; }
  cosb[t] = d / fmaxf(sqrtf(na)*sqrtf(nb), 1e-8f);
}
// top-5 per row + gather hard-negative embeddings
__global__ void k_top5(const float* __restrict__ cosb, const int* __restrict__ noise,
                       const float* __restrict__ v_emb, int* __restrict__ neg_ids,
                       float* __restrict__ emb_neg_v)
{
  const int b = blockIdx.x;
  __shared__ float vals[512];
  __shared__ float rv[256]; __shared__ int ri[256]; __shared__ int sel;
  vals[threadIdx.x]       = cosb[(size_t)b*512 + threadIdx.x];
  vals[threadIdx.x + 256] = cosb[(size_t)b*512 + 256 + threadIdx.x];
  __syncthreads();
  for (int k = 0; k < NEG; ++k) {
    float bv = -3.0e38f; int bi = 1 << 30;
    for (int q = threadIdx.x; q < 512; q += 256) {
      float v = vals[q];
      if (v > bv || (v == bv && q < bi)) { bv = v; bi = q; }
    }
    rv[threadIdx.x] = bv; ri[threadIdx.x] = bi; __syncthreads();
    for (int off = 128; off > 0; off >>= 1) {
      if (threadIdx.x < off) {
        float ov = rv[threadIdx.x+off]; int oi = ri[threadIdx.x+off];
        if (ov > rv[threadIdx.x] || (ov == rv[threadIdx.x] && oi < ri[threadIdx.x]))
          { rv[threadIdx.x] = ov; ri[threadIdx.x] = oi; }
      }
      __syncthreads();
    }
    if (threadIdx.x == 0) {
      sel = ri[0];
      neg_ids[b*NEG + k] = noise[b*NOISE + ri[0]];
      vals[ri[0]] = -3.0e38f;
    }
    __syncthreads();
    int nid = noise[b*NOISE + sel];
    if (threadIdx.x < E)
      emb_neg_v[((size_t)b*NEG + k)*E + threadIdx.x] = v_emb[(size_t)nid*E + threadIdx.x];
    __syncthreads();
  }
}
__global__ void k_build_ids(const int* __restrict__ pos_v, const int* __restrict__ neg_ids,
                            int* __restrict__ ids)
{
  int t = blockIdx.x*256 + threadIdx.x; if (t >= NW) return;
  ids[t] = (t < BATCH*W) ? pos_v[t] : neg_ids[t - BATCH*W];
}
// x[T, NW, E] f16, time-major
__global__ void k_gather_x(const int* __restrict__ ids, const int* __restrict__ w2c,
                           const _Float16* __restrict__ img_emb, _Float16* __restrict__ x)
{
  size_t t = (size_t)blockIdx.x*256 + threadIdx.x;
  if (t >= (size_t)NW*CPW*E) return;
  int e = (int)(t & 127); size_t rest = t >> 7;
  int tt = (int)(rest % CPW); int i = (int)(rest / CPW);
  int ch = w2c[(size_t)ids[i]*CPW + tt];
  x[((size_t)tt*NW + i)*E + e] = img_emb[(size_t)ch*E + e];
}

// ============================================================ LSTM cell
__global__ void k_lstm_cell(const float* __restrict__ xw, const float* __restrict__ hw,
                            float* __restrict__ cst, _Float16* __restrict__ h,
                            _Float16* __restrict__ outc, int col0)
{
  size_t t = (size_t)blockIdx.x*256 + threadIdx.x;
  if (t >= (size_t)NW*E) return;
  int j = (int)(t & 127); size_t row = t >> 7;
  const float* xr = xw + row*512;
  const float* hr = hw + row*512;
  float gi = xr[j]       + hr[j];
  float gf = xr[128 + j] + hr[128 + j];
  float gg = xr[256 + j] + hr[256 + j];
  float go = xr[384 + j] + hr[384 + j];
  float cc = sigmf(gf) * cst[t] + sigmf(gi) * tanhf(gg);
  float hh = sigmf(go) * tanhf(cc);
  cst[t] = cc;
  h[t] = (_Float16)hh;
  outc[row*256 + col0 + j] = (_Float16)hh;
}

// ============================================================ attention
__global__ void k_att_score(const float* __restrict__ attA, const float* __restrict__ a2w,
                            const float* __restrict__ a2b, float* __restrict__ s)
{
  int t = blockIdx.x*256 + threadIdx.x; if (t >= TN) return;
  float acc = a2b[0];
  const float* r = attA + (size_t)t*E;
  for (int e = 0; e < E; ++e) acc += r[e] * a2w[e];
  s[t] = acc;
}
__global__ void k_att_softmax(const float* __restrict__ s, float* __restrict__ alpha)
{
  int i = blockIdx.x*256 + threadIdx.x; if (i >= NW) return;
  float v[CPW], mx = -3.0e38f;
#pragma unroll
  for (int k = 0; k < CPW; ++k) { v[k] = s[(size_t)k*NW + i]; mx = fmaxf(mx, v[k]); }
  float sum = 0.f;
#pragma unroll
  for (int k = 0; k < CPW; ++k) { v[k] = expf(v[k] - mx); sum += v[k]; }
#pragma unroll
  for (int k = 0; k < CPW; ++k) alpha[(size_t)k*NW + i] = v[k] / sum;
}
__global__ void k_att_wsum(const float* __restrict__ alpha, const _Float16* __restrict__ outc,
                           _Float16* __restrict__ y)
{
  size_t t = (size_t)blockIdx.x*256 + threadIdx.x;
  if (t >= (size_t)NW*256) return;
  int col = (int)(t & 255); size_t i = t >> 8;
  float s = 0.f;
#pragma unroll
  for (int k = 0; k < CPW; ++k)
    s += alpha[(size_t)k*NW + i] * (float)outc[((size_t)k*NW + i)*256 + col];
  y[t] = (_Float16)s;
}

// ============================================================ loss
__global__ void k_loss(const float* __restrict__ emb_u, const float* __restrict__ emb_v,
                       const float* __restrict__ emb_char, const float* __restrict__ emb_neg_v,
                       float* __restrict__ lossb)
{
  int b = blockIdx.x*256 + threadIdx.x; if (b >= BATCH) return;
  const float* eu = emb_u + (size_t)b*E;
  float loss = 0.f, d;
  // context char score (mean over W)
  d = 0.f;
  for (int e = 0; e < E; ++e) {
    float m = 0.f;
    for (int w = 0; w < W; ++w) m += emb_char[((size_t)b*W + w)*E + e];
    d += eu[e] * m * (1.f/6.f);
  }
  loss += log1pf(expf(-clip10(d)));
  // negative char scores
  for (int n = 0; n < NEG; ++n) {
    d = 0.f;
    const float* r = emb_char + ((size_t)(BATCH*W) + (size_t)b*NEG + n)*E;
    for (int e = 0; e < E; ++e) d += r[e] * eu[e];
    loss += log1pf(expf(clip10(d)));
  }
  // word score
  d = 0.f;
  for (int e = 0; e < E; ++e) d += eu[e] * emb_v[(size_t)b*E + e];
  loss += log1pf(expf(-clip10(d)));
  // negative word scores
  for (int n = 0; n < NEG; ++n) {
    d = 0.f;
    const float* r = emb_neg_v + ((size_t)b*NEG + n)*E;
    for (int e = 0; e < E; ++e) d += r[e] * eu[e];
    loss += log1pf(expf(clip10(d)));
  }
  lossb[b] = loss;
}
__global__ void k_loss_reduce(const float* __restrict__ lossb, float* __restrict__ out)
{
  __shared__ float r[256];
  float s = lossb[threadIdx.x] + lossb[threadIdx.x+256] + lossb[threadIdx.x+512] + lossb[threadIdx.x+768];
  r[threadIdx.x] = s; __syncthreads();
  for (int off = 128; off > 0; off >>= 1) {
    if (threadIdx.x < off) r[threadIdx.x] += r[threadIdx.x+off];
    __syncthreads();
  }
  if (threadIdx.x == 0) out[0] = r[0] / (float)BATCH;
}

// ============================================================ launch
extern "C" void kernel_launch(void* const* d_in, const int* in_sizes, int n_in,
                              void* d_out, int out_size, void* d_ws, size_t ws_size,
                              hipStream_t stream) {
  (void)in_sizes; (void)n_in; (void)out_size; (void)ws_size;
  const int*   pos_u  = (const int*)  d_in[0];
  const int*   pos_v  = (const int*)  d_in[1];
  const int*   noise  = (const int*)  d_in[2];
  const int*   w2c    = (const int*)  d_in[3];
  const float* img    = (const float*)d_in[4];
  const float* u_emb  = (const float*)d_in[5];
  const float* v_emb  = (const float*)d_in[6];
  const float* c1w = (const float*)d_in[7],  *c1b = (const float*)d_in[8];
  const float* g1  = (const float*)d_in[9],  *b1  = (const float*)d_in[10];
  const float* c2w = (const float*)d_in[11], *c2b = (const float*)d_in[12];
  const float* g2  = (const float*)d_in[13], *b2  = (const float*)d_in[14];
  const float* fcw = (const float*)d_in[15], *fcb = (const float*)d_in[16];
  const float* g3  = (const float*)d_in[17], *b3  = (const float*)d_in[18];
  const float* wi_f = (const float*)d_in[19], *wh_f = (const float*)d_in[20];
  const float* bi_f = (const float*)d_in[21], *bh_f = (const float*)d_in[22];
  const float* wi_b = (const float*)d_in[23], *wh_b = (const float*)d_in[24];
  const float* bi_b = (const float*)d_in[25], *bh_b = (const float*)d_in[26];
  const float* a1w = (const float*)d_in[27], *a1b = (const float*)d_in[28];
  const float* a2w = (const float*)d_in[29], *a2b = (const float*)d_in[30];
  const float* a3w = (const float*)d_in[31], *a3b = (const float*)d_in[32];

  char* ws = (char*)d_ws;
  size_t off = 0;
  auto alloc = [&](size_t bytes)->char* {
    size_t p = (off + 255) & ~(size_t)255; off = p + bytes; return ws + p;
  };
  float*     stats   = (float*)alloc(256*4);          // [0..127]=sum, [128..255]=sumsq
  float*     stA = stats, *stB = stats + 128;
  float*     bsum_f  = (float*)alloc(512*4);
  float*     bsum_b  = (float*)alloc(512*4);
  int*       neg_ids = (int*)  alloc(BATCH*NEG*4);
  int*       ids     = (int*)  alloc(NW*4);
  float*     lossb   = (float*)alloc(BATCH*4);
  _Float16*  fcw_h   = (_Float16*)alloc((size_t)128*2048*2);
  _Float16*  wif_h   = (_Float16*)alloc((size_t)512*128*2);
  _Float16*  whf_h   = (_Float16*)alloc((size_t)512*128*2);
  _Float16*  wib_h   = (_Float16*)alloc((size_t)512*128*2);
  _Float16*  whb_h   = (_Float16*)alloc((size_t)512*128*2);
  _Float16*  a1w_h   = (_Float16*)alloc((size_t)128*256*2);
  _Float16*  a3w_h   = (_Float16*)alloc((size_t)128*256*2);
  _Float16*  c2w_h   = (_Float16*)alloc((size_t)32*288*2);
  _Float16*  imgemb_h= (_Float16*)alloc((size_t)CHARS*E*2);
  float*     emb_u   = (float*)alloc((size_t)BATCH*E*4);
  float*     emb_v   = (float*)alloc((size_t)BATCH*E*4);
  float*     emb_neg_v=(float*)alloc((size_t)BATCH*NEG*E*4);
  float*     cosb    = (float*)alloc((size_t)BATCH*NOISE*4);
  // aliased big arena: CNN-phase layout overlaps LSTM-gate layout (disjoint lifetimes)
  char*      R1      = alloc(230686720ULL);
  float*     pool1   = (float*)R1;                          // 84,934,656 B
  float*     conv2o  = (float*)(R1 + 84934656ULL);          // 67,108,864 B
  _Float16*  pool2_h = (_Float16*)(R1 + 152043520ULL);      //  8,388,608 B
  float*     fcout   = (float*)(R1 + 160432128ULL);         //  1,048,576 B
  float*     xw_f    = (float*)R1;                          // 115,343,360 B
  float*     xw_b    = (float*)(R1 + 115343360ULL);
  _Float16*  x_h     = (_Float16*)alloc((size_t)TN*E*2);
  _Float16*  out_c   = (_Float16*)alloc((size_t)TN*256*2);
  _Float16*  h_h     = (_Float16*)alloc((size_t)NW*E*2);
  float*     cst     = (float*)alloc((size_t)NW*E*4);
  float*     gbuf    = (float*)alloc((size_t)NW*512*4);
  float*     attA    = (float*)alloc((size_t)TN*E*4);
  float*     satt    = (float*)alloc((size_t)TN*4);
  float*     alpha   = (float*)alloc((size_t)TN*4);
  _Float16*  y_h     = (_Float16*)alloc((size_t)NW*256*2);
  float*     emb_char= (float*)alloc((size_t)NW*E*4);

  const dim3 B256(256);
  auto blks = [](size_t n){ return dim3((unsigned)((n + 255)/256)); };

  // weight casts + bias sums
  k_cast_f16<<<blks(262144),B256,0,stream>>>(fcw, fcw_h, 262144);
  k_cast_f16<<<blks(65536), B256,0,stream>>>(wi_f, wif_h, 65536);
  k_cast_f16<<<blks(65536), B256,0,stream>>>(wh_f, whf_h, 65536);
  k_cast_f16<<<blks(65536), B256,0,stream>>>(wi_b, wib_h, 65536);
  k_cast_f16<<<blks(65536), B256,0,stream>>>(wh_b, whb_h, 65536);
  k_cast_f16<<<blks(32768), B256,0,stream>>>(a1w, a1w_h, 32768);
  k_cast_f16<<<blks(32768), B256,0,stream>>>(a3w, a3w_h, 32768);
  k_cast_f16<<<blks(9216),  B256,0,stream>>>(c2w, c2w_h, 9216);
  k_vecadd<<<blks(512),B256,0,stream>>>(bi_f, bh_f, bsum_f, 512);
  k_vecadd<<<blks(512),B256,0,stream>>>(bi_b, bh_b, bsum_b, 512);

  // ---- CNN
  k_zero_f32<<<1,B256,0,stream>>>(stats, 256);
  k_conv1_stats<<<65536,B256,0,stream>>>(img, c1w, c1b, stA, stB);
  k_bn1_pool<<<blks((size_t)2048*32*18*18),B256,0,stream>>>(img, c1w, c1b, stA, stB, g1, b1, pool1);
  k_conv2_wmma<<<32768,32,0,stream>>>(pool1, c2w_h, c2b, conv2o);
  k_zero_f32<<<1,B256,0,stream>>>(stats, 256);
  k_plane_stats256<<<65536,B256,0,stream>>>(conv2o, stA, stB);
  k_bn2_pool<<<blks((size_t)2048*32*8*8),B256,0,stream>>>(conv2o, stA, stB, g2, b2, pool2_h);
  k_gemm_wmma<<<dim3(128,2),32,0,stream>>>(pool2_h,2048, fcw_h,2048, fcout,128, 2048, fcb, 0);
  k_bn3_stats<<<128,B256,0,stream>>>(fcout, stA, stB);
  k_bn3_relu_f16<<<blks(262144),B256,0,stream>>>(fcout, stA, stB, g3, b3, imgemb_h);

  // ---- embeddings + hard-negative mining
  k_gather_u<<<blks(BATCH*E),B256,0,stream>>>(u_emb, pos_u, emb_u);
  k_gather_v_mean<<<blks(BATCH*E),B256,0,stream>>>(v_emb, pos_v, emb_v);
  k_cos<<<blks((size_t)BATCH*NOISE),B256,0,stream>>>(emb_v, v_emb, noise, cosb);
  k_top5<<<BATCH,B256,0,stream>>>(cosb, noise, v_emb, neg_ids, emb_neg_v);
  k_build_ids<<<blks(NW),B256,0,stream>>>(pos_v, neg_ids, ids);
  k_gather_x<<<blks((size_t)NW*CPW*E),B256,0,stream>>>(ids, w2c, imgemb_h, x_h);

  // ---- BiLSTM (input transforms, then per-step recurrent GEMM + cell)
  k_gemm_wmma<<<dim3(TN/16,8),32,0,stream>>>(x_h,128, wif_h,128, xw_f,512, 128, bsum_f, 0);
  k_gemm_wmma<<<dim3(TN/16,8),32,0,stream>>>(x_h,128, wib_h,128, xw_b,512, 128, bsum_b, 0);
  // forward
  k_zero_f16<<<blks((size_t)NW*E),B256,0,stream>>>(h_h, NW*E);
  k_zero_f32<<<blks((size_t)NW*E),B256,0,stream>>>(cst, NW*E);
  for (int t = 0; t < CPW; ++t) {
    k_gemm_wmma<<<dim3(NW/16,8),32,0,stream>>>(h_h,128, whf_h,128, gbuf,512, 128, nullptr, 0);
    k_lstm_cell<<<blks((size_t)NW*E),B256,0,stream>>>(xw_f + (size_t)t*NW*512, gbuf, cst, h_h,
                                                      out_c + (size_t)t*NW*256, 0);
  }
  // backward (processes reversed time, stores at original index)
  k_zero_f16<<<blks((size_t)NW*E),B256,0,stream>>>(h_h, NW*E);
  k_zero_f32<<<blks((size_t)NW*E),B256,0,stream>>>(cst, NW*E);
  for (int t = CPW-1; t >= 0; --t) {
    k_gemm_wmma<<<dim3(NW/16,8),32,0,stream>>>(h_h,128, whb_h,128, gbuf,512, 128, nullptr, 0);
    k_lstm_cell<<<blks((size_t)NW*E),B256,0,stream>>>(xw_b + (size_t)t*NW*512, gbuf, cst, h_h,
                                                      out_c + (size_t)t*NW*256, 128);
  }

  // ---- attention + projection
  k_gemm_wmma<<<dim3(TN/16,2),32,0,stream>>>(out_c,256, a1w_h,256, attA,128, 256, a1b, 2);
  k_att_score<<<blks(TN),B256,0,stream>>>(attA, a2w, a2b, satt);
  k_att_softmax<<<blks(NW),B256,0,stream>>>(satt, alpha);
  k_att_wsum<<<blks((size_t)NW*256),B256,0,stream>>>(alpha, out_c, y_h);
  k_gemm_wmma<<<dim3(NW/16,2),32,0,stream>>>(y_h,256, a3w_h,256, emb_char,128, 256, a3b, 0);

  // ---- loss
  k_loss<<<blks(BATCH),B256,0,stream>>>(emb_u, emb_v, emb_char, emb_neg_v, lossb);
  k_loss_reduce<<<1,B256,0,stream>>>(lossb, (float*)d_out);
}
